// SurvivalMoE_56384330662353
// MI455X (gfx1250) — compile-verified, matrix operand order checked by reading, not run on hardware
//
#include <hip/hip_runtime.h>
#include <hip/hip_bf16.h>
#include <math.h>

// ---------------------------------------------------------------------------
// SurvivalMoE for MI455X (gfx1250, wave32, WMMA bf16 16x16x32)
// ---------------------------------------------------------------------------
typedef __attribute__((ext_vector_type(8)))  __bf16 v8bf;
typedef __attribute__((ext_vector_type(16))) __bf16 v16bf;
typedef __attribute__((ext_vector_type(8)))  float  v8f;

#define B_ 8192
#define D_ 512
#define H_ 2048
#define E_ 3
#define N_ 256
#define S_ 5

// One hash -> two Bernoulli(0.9) keep decisions (bits [15:0] and [31:16]).
static __device__ __forceinline__ uint32_t hash_u32(uint32_t x) {
  x ^= x >> 16; x *= 0x7feb352dU;
  x ^= x >> 15; x *= 0x846ca68bU;
  x ^= x >> 16;
  return x;
}
#define KEEP_THRESH 58982u  // 0.9 * 65536

// Load a 16x32 bf16 tile (row-major, row stride `ld` elements) into the WMMA
// A/B lane layout: lane<16 -> K=[0..7]+[16..23], lane>=16 -> K=[8..15]+[24..31]
static __device__ __forceinline__ v16bf load_tile_k32(const __bf16* tile, int ld, int lane) {
  const int r  = lane & 15;
  const int kb = (lane < 16) ? 0 : 8;
  const __bf16* p = tile + (size_t)r * ld + kb;
  v8bf lo = *reinterpret_cast<const v8bf*>(p);
  v8bf hi = *reinterpret_cast<const v8bf*>(p + 16);
  v16bf o;
#pragma unroll
  for (int j = 0; j < 8; ++j) { o[j] = lo[j]; o[8 + j] = hi[j]; }
  return o;
}

static __device__ __forceinline__ v8f wmma_bf16(v16bf a, v16bf b, v8f c) {
  return __builtin_amdgcn_wmma_f32_16x16x32_bf16(
      /*neg_a=*/false, a, /*neg_b=*/false, b,
      /*c_mod=*/(short)0, c, /*reuse_a=*/false, /*reuse_b=*/false);
}

// ---------------------------------------------------------------------------
// K0: fp32 -> bf16 convert (grid-stride)
// ---------------------------------------------------------------------------
__global__ void SMoE_cvt_bf16(const float* __restrict__ src,
                              __bf16* __restrict__ dst, int n) {
  for (int i = blockIdx.x * blockDim.x + threadIdx.x; i < n;
       i += gridDim.x * blockDim.x)
    dst[i] = (__bf16)src[i];
}

// ---------------------------------------------------------------------------
// A1: per expert e:  G = GELU(LN(z @ w1[e]^T + b1[e])) * (1/0.9)   (bf16)
// Block = 1024 threads (32 waves); block covers 16 rows of B x full H=2048.
// Wave w owns 4 N-tiles (h0 = w*64): low acc pressure (32 VGPRs) so all
// 5 tile loads of a k-step stay in distinct registers ahead of the 4 WMMAs
// (hint: 10 VMEM reads, then 4 WMMAs -> staggered loadcnt waits).
// ---------------------------------------------------------------------------
__global__ __launch_bounds__(1024) void SMoE_predict_gelu(
    int e, const __bf16* __restrict__ zbf, const __bf16* __restrict__ w1bf,
    const float* __restrict__ b1, const float* __restrict__ ln_g,
    const float* __restrict__ ln_b, __bf16* __restrict__ G) {
  __shared__ float ssum[16], ssq[16], smean[16], srstd[16];
  const int tid  = threadIdx.x;
  const int wv   = tid >> 5;
  const int lane = tid & 31;
  const int b0   = blockIdx.x * 16;
  const int h0   = wv * 64;
  const int col_lane = lane & 15;
  const int half = (lane < 16) ? 0 : 1;

  if (tid < 16) { ssum[tid] = 0.f; ssq[tid] = 0.f; }
  __syncthreads();

  v8f acc[4] = {};
  const __bf16* ztile = zbf + (size_t)b0 * D_;
  const __bf16* wbase = w1bf + (size_t)e * H_ * D_ + (size_t)h0 * D_;
  for (int k = 0; k < D_ / 32; ++k) {
    v16bf a = load_tile_k32(ztile + k * 32, D_, lane);
    v16bf bt[4];
#pragma unroll
    for (int t = 0; t < 4; ++t)
      bt[t] = load_tile_k32(wbase + (size_t)t * 16 * D_ + k * 32, D_, lane);
#pragma unroll
    for (int t = 0; t < 4; ++t)
      acc[t] = wmma_bf16(a, bt[t], acc[t]);
    // pin all 10 b128 reads ahead of the 4-WMMA burst
    __builtin_amdgcn_sched_group_barrier(0x020, 10, 0);  // VMEM read
    __builtin_amdgcn_sched_group_barrier(0x008, 4, 0);   // MFMA/WMMA
  }

  // bias + LN partial stats (rows m = half*8 + i)
  float ps[8] = {}, pq[8] = {};
#pragma unroll
  for (int t = 0; t < 4; ++t) {
    const int col = h0 + t * 16 + col_lane;
    const float bias = b1[e * H_ + col];
#pragma unroll
    for (int i = 0; i < 8; ++i) {
      float v = acc[t][i] + bias;
      acc[t][i] = v;
      ps[i] += v; pq[i] += v * v;
    }
  }
#pragma unroll
  for (int i = 0; i < 8; ++i) {
    float s = ps[i], q = pq[i];
    s += __shfl_xor(s, 1, 16); q += __shfl_xor(q, 1, 16);
    s += __shfl_xor(s, 2, 16); q += __shfl_xor(q, 2, 16);
    s += __shfl_xor(s, 4, 16); q += __shfl_xor(q, 4, 16);
    s += __shfl_xor(s, 8, 16); q += __shfl_xor(q, 8, 16);
    if (col_lane == 0) {
      atomicAdd(&ssum[half * 8 + i], s);
      atomicAdd(&ssq [half * 8 + i], q);
    }
  }
  __syncthreads();
  if (tid < 16) {
    float mu  = ssum[tid] * (1.0f / H_);
    float var = ssq[tid] * (1.0f / H_) - mu * mu;   // ddof=0 (jnp.var)
    smean[tid] = mu;
    srstd[tid] = rsqrtf(var + 1e-5f);
  }
  __syncthreads();

  // normalize + exact GELU + fold dropout scale, store bf16
#pragma unroll
  for (int t = 0; t < 4; ++t) {
    const int col = h0 + t * 16 + col_lane;
    const float gam = ln_g[e * H_ + col];
    const float bet = ln_b[e * H_ + col];
#pragma unroll
    for (int i = 0; i < 8; ++i) {
      const int m = half * 8 + i;
      float v  = (acc[t][i] - smean[m]) * srstd[m] * gam + bet;
      float ge = 0.5f * v * (1.0f + erff(v * 0.70710678118654752f));
      G[(size_t)(b0 + m) * H_ + col] = (__bf16)(ge * (1.0f / 0.9f));
    }
  }
}

// ---------------------------------------------------------------------------
// A2: per expert e: preds_s = mask_s(G) @ w2[e]^T ; ent[b,e] =
//   mean_d Var_s(preds) (ddof=1).  b2 dropped (constant over s -> no var).
// Block = 1024 threads (32 waves); 16 rows of B x D=512, wave w owns 1
// N-tile (d0 = w*16).  k outer / s inner: tiles loaded once per k for all
// 5 samples; k-loop software-pipelined (cur/nxt rotation, last iter peeled)
// with hints pinning the 4 k+1 loads ahead of the 5-WMMA + mask-VALU body.
// ---------------------------------------------------------------------------
__global__ __launch_bounds__(1024) void SMoE_mc_entropy(
    int e, const __bf16* __restrict__ G, const __bf16* __restrict__ w2bf,
    float* __restrict__ ent) {
  __shared__ float sent[16];
  const int tid  = threadIdx.x;
  const int wv   = tid >> 5;
  const int lane = tid & 31;
  const int b0   = blockIdx.x * 16;
  const int d0   = wv * 16;
  const int r    = lane & 15;
  const int kb   = (lane < 16) ? 0 : 8;
  const int half = (lane < 16) ? 0 : 1;

  if (tid < 16) sent[tid] = 0.f;
  __syncthreads();

  v8f acc[S_] = {};
  const __bf16* w2tile = w2bf + (size_t)e * D_ * H_ + (size_t)d0 * H_;
  const __bf16* grow   = G + (size_t)(b0 + r) * H_ + kb;
  const uint32_t mbase0 = ((uint32_t)e * S_ * B_ + (uint32_t)(b0 + r)) * H_ + (uint32_t)kb;

  // per-k compute: build 5 masked A variants, 5 WMMAs
  auto compute = [&](v8bf lo, v8bf hi, v16bf bt, int k) {
#pragma unroll
    for (int s = 0; s < S_; ++s) {
      const uint32_t base = mbase0 + (uint32_t)s * (B_ * H_) + (uint32_t)(k * 32);
      v16bf a;
#pragma unroll
      for (int j = 0; j < 8; ++j) {
        uint32_t hx = hash_u32(base + j);          // 2 decisions per hash
        a[j]     = ((hx & 0xFFFFu) < KEEP_THRESH) ? lo[j] : (__bf16)0.0f;
        a[8 + j] = ((hx >> 16)     < KEEP_THRESH) ? hi[j] : (__bf16)0.0f;
      }
      acc[s] = wmma_bf16(a, bt, acc[s]);
    }
  };

  // prologue: load tiles for k = 0
  v8bf  lo_c = *reinterpret_cast<const v8bf*>(grow);
  v8bf  hi_c = *reinterpret_cast<const v8bf*>(grow + 16);
  v16bf bt_c = load_tile_k32(w2tile, H_, lane);

  for (int k = 0; k < H_ / 32 - 1; ++k) {
    // issue loads for k+1 before consuming k's tiles
    const __bf16* pn = grow + (k + 1) * 32;
    v8bf  lo_n = *reinterpret_cast<const v8bf*>(pn);
    v8bf  hi_n = *reinterpret_cast<const v8bf*>(pn + 16);
    v16bf bt_n = load_tile_k32(w2tile + (k + 1) * 32, H_, lane);

    compute(lo_c, hi_c, bt_c, k);

    // k+1 loads (4 b128/b64 reads) ahead of the 5-WMMA burst
    __builtin_amdgcn_sched_group_barrier(0x020, 4, 0);  // VMEM read
    __builtin_amdgcn_sched_group_barrier(0x008, 5, 0);  // MFMA/WMMA

    lo_c = lo_n; hi_c = hi_n; bt_c = bt_n;
  }
  compute(lo_c, hi_c, bt_c, H_ / 32 - 1);   // epilogue

  // per-(b,d) unbiased variance over S, then reduce over d
  float pv[8];
#pragma unroll
  for (int i = 0; i < 8; ++i) {
    float sm = 0.f, qq = 0.f;
#pragma unroll
    for (int s = 0; s < S_; ++s) {
      float v = acc[s][i];
      sm += v; qq += v * v;
    }
    pv[i] = (qq - sm * sm * (1.0f / S_)) * (1.0f / (S_ - 1));
  }
#pragma unroll
  for (int i = 0; i < 8; ++i) {
    float v = pv[i];
    v += __shfl_xor(v, 1, 16);
    v += __shfl_xor(v, 2, 16);
    v += __shfl_xor(v, 4, 16);
    v += __shfl_xor(v, 8, 16);
    if ((lane & 15) == 0) atomicAdd(&sent[half * 8 + i], v);
  }
  __syncthreads();
  if (tid < 16)
    ent[(size_t)(b0 + tid) * E_ + e] = sent[tid] * (1.0f / D_);
}

// ---------------------------------------------------------------------------
// B1: per-row argmin expert + survival dispatch.
// Block = 256 threads = one row b. Gate dot products from LDS-staged z row;
// output = uniform-branch sum of active eow rows (h_active is 0/1).
// ---------------------------------------------------------------------------
__global__ __launch_bounds__(256) void SMoE_dispatch(
    const float* __restrict__ z, const float* __restrict__ ew,
    const float* __restrict__ eb, const float* __restrict__ eow,
    const float* __restrict__ dorm, const float* __restrict__ ent,
    float* __restrict__ out) {
  __shared__ float zsh[D_];
  __shared__ float flag[N_];
  __shared__ int esel_sh;
  const int b = blockIdx.x;
  const int tid = threadIdx.x;

  if (tid == 0) {
    float e0 = ent[(size_t)b * E_ + 0];
    float e1 = ent[(size_t)b * E_ + 1];
    float e2 = ent[(size_t)b * E_ + 2];
    int sel = 0; float m = e0;
    if (e1 < m) { m = e1; sel = 1; }
    if (e2 < m) { m = e2; sel = 2; }
    esel_sh = sel;
  }
  zsh[tid]       = z[(size_t)b * D_ + tid];
  zsh[256 + tid] = z[(size_t)b * D_ + 256 + tid];
  __syncthreads();
  const int e = esel_sh;

  // gate: h = z . ew[e][n] + eb ; active = (h > 0) && (dormancy <= 30)
  const float4* wrow4 =
      reinterpret_cast<const float4*>(ew + ((size_t)e * N_ + tid) * D_);
  float acc = eb[e * N_ + tid];
#pragma unroll 4
  for (int d4 = 0; d4 < D_ / 4; ++d4) {
    float4 w4 = wrow4[d4];
    acc += zsh[d4 * 4 + 0] * w4.x + zsh[d4 * 4 + 1] * w4.y +
           zsh[d4 * 4 + 2] * w4.z + zsh[d4 * 4 + 3] * w4.w;
  }
  flag[tid] = (acc > 0.f && dorm[e * N_ + tid] <= 30.0f) ? 1.f : 0.f;
  __syncthreads();

  float o0 = 0.f, o1 = 0.f;
  const float* eo = eow + (size_t)e * N_ * D_;
  for (int n = 0; n < N_; ++n) {
    if (flag[n] != 0.f) {           // uniform across the block
      o0 += eo[(size_t)n * D_ + tid];
      o1 += eo[(size_t)n * D_ + 256 + tid];
    }
  }
  out[(size_t)b * D_ + tid]       = o0;
  out[(size_t)b * D_ + 256 + tid] = o1;
}

// ---------------------------------------------------------------------------
extern "C" void kernel_launch(void* const* d_in, const int* in_sizes, int n_in,
                              void* d_out, int out_size, void* d_ws, size_t ws_size,
                              hipStream_t stream) {
  (void)in_sizes; (void)n_in; (void)out_size; (void)ws_size;
  const float* z    = (const float*)d_in[0];
  const float* w1   = (const float*)d_in[1];
  const float* b1   = (const float*)d_in[2];
  const float* ln_g = (const float*)d_in[3];
  const float* ln_b = (const float*)d_in[4];
  const float* w2   = (const float*)d_in[5];
  // d_in[6] = b2: constant over MC samples -> no effect on variance; unused.
  const float* ew   = (const float*)d_in[7];
  const float* eb   = (const float*)d_in[8];
  const float* eow  = (const float*)d_in[9];
  const float* dorm = (const float*)d_in[10];
  float* out = (float*)d_out;

  char* ws = (char*)d_ws;
  size_t off = 0;
  auto alloc = [&](size_t bytes) -> char* {
    char* p = ws + off;
    off = (off + bytes + 255) & ~(size_t)255;
    return p;
  };
  __bf16* zbf  = (__bf16*)alloc((size_t)B_ * D_ * 2);        //  8.0 MB
  __bf16* w1bf = (__bf16*)alloc((size_t)E_ * H_ * D_ * 2);   //  6.0 MB
  __bf16* w2bf = (__bf16*)alloc((size_t)E_ * D_ * H_ * 2);   //  6.0 MB
  __bf16* G    = (__bf16*)alloc((size_t)B_ * H_ * 2);        // 32.0 MB (reused per expert)
  float*  ent  = (float*)alloc((size_t)B_ * E_ * 4);         //  0.1 MB

  // Phase 0: bf16 conversions (weights are L2-resident thereafter)
  {
    const int nz = B_ * D_, nw = E_ * H_ * D_;
    SMoE_cvt_bf16<<<(nz + 255) / 256, 256, 0, stream>>>(z,  zbf,  nz);
    SMoE_cvt_bf16<<<(nw + 255) / 256, 256, 0, stream>>>(w1, w1bf, nw);
    SMoE_cvt_bf16<<<(nw + 255) / 256, 256, 0, stream>>>(w2, w2bf, nw);
  }

  // Phase A: router (G buffer reused per expert; stream order serializes)
  for (int e = 0; e < E_; ++e) {
    SMoE_predict_gelu<<<B_ / 16, 1024, 0, stream>>>(e, zbf, w1bf, b1, ln_g, ln_b, G);
    SMoE_mc_entropy  <<<B_ / 16, 1024, 0, stream>>>(e, G, w2bf, ent);
  }

  // Phase B: argmin + survival-expert dispatch (selected expert only)
  SMoE_dispatch<<<B_, 256, 0, stream>>>(z, ew, eb, eow, dorm, ent, out);
}
